// MultiHeadAttention_2224793059797
// MI455X (gfx1250) — compile-verified, matrix operand order checked by reading, not run on hardware
//
#include <hip/hip_runtime.h>

typedef __attribute__((ext_vector_type(16))) __bf16          bf16x16;
typedef __attribute__((ext_vector_type(8)))  float           v8f;
typedef __attribute__((ext_vector_type(8)))  unsigned short  u16x8;
typedef unsigned short u16;

#define B_  4
#define T_  2048
#define C_  1024
#define H_  16
#define D_  64
#define M_  (B_ * T_)    /* 8192 */
#define N3_ (3 * C_)     /* 3072 */
#define LDSS 72          /* LDS row stride (bf16 elems): 144B, 16B aligned, conflict-free */

// ---------------------------------------------------------------- helpers ---

static __device__ __forceinline__ u16 f32_to_bf16(float f) {
  union { float f; unsigned u; } x; x.f = f;
  unsigned r = x.u + 0x7FFFu + ((x.u >> 16) & 1u);   // round-to-nearest-even
  return (u16)(r >> 16);
}

static __device__ __forceinline__ v8f wmma_bf16(bf16x16 a, bf16x16 b, v8f c) {
  return __builtin_amdgcn_wmma_f32_16x16x32_bf16(false, a, false, b, (short)0, c,
                                                 false, false);
}

// Async 16B copy global -> LDS (ASYNCcnt-tracked, per-lane).
static __device__ __forceinline__ void async_copy16(u16* dst_lds, const u16* src) {
  unsigned lds_off = (unsigned)(unsigned long long)dst_lds;   // low 32b = LDS byte addr
  unsigned long long ga = (unsigned long long)src;
  asm volatile("global_load_async_to_lds_b128 %0, %1, off"
               :: "v"(lds_off), "v"(ga) : "memory");
}
static __device__ __forceinline__ void wait_async() {
  asm volatile("s_wait_asynccnt 0x0" ::: "memory");
}

// Load a 16x32 A-fragment (or B^T fragment, same lane layout) from LDS.
// Lane 0-15 : row = base+lane,    K = {0..7, 16..23}
// Lane 16-31: row = base+lane-16, K = {8..15, 24..31}
static __device__ __forceinline__ bf16x16 load_frag(const u16* smem, int row_base,
                                                    int k_base, int lane) {
  int r   = row_base + (lane & 15);
  int klo = k_base + ((lane >> 4) << 3);
  u16x8 lo = *reinterpret_cast<const u16x8*>(smem + r * LDSS + klo);
  u16x8 hi = *reinterpret_cast<const u16x8*>(smem + r * LDSS + klo + 16);
  union { bf16x16 v; u16 u[16]; } t;
#pragma unroll
  for (int i = 0; i < 8; ++i) { t.u[i] = lo[i]; t.u[8 + i] = hi[i]; }
  return t.v;
}

// Cooperative async stage of a (rows x 64) bf16 tile into LDS (stride LDSS).
static __device__ __forceinline__ void stage_tile64(u16* dst, const u16* src, int rows,
                                                    int gstride, int tid, int nthr) {
  int chunks = rows * 8;                       // 8 x 16B chunks per row
  for (int i = tid; i < chunks; i += nthr) {
    int r = i >> 3, c = (i & 7) << 3;
    async_copy16(dst + r * LDSS + c, src + r * gstride + c);
  }
}

// One K=64 compute step: 24 ds_load_b128 -> 16 back-to-back WMMAs per wave.
static __device__ __forceinline__ void gemm_compute(const u16* sA, const u16* sB,
                                                    v8f acc0[4], v8f acc1[4],
                                                    int lane, int wave) {
#pragma unroll
  for (int kk = 0; kk < 2; ++kk) {
    bf16x16 a0 = load_frag(sA, wave * 32, kk * 32, lane);
    bf16x16 a1 = load_frag(sA, wave * 32 + 16, kk * 32, lane);
    bf16x16 bf[4];
#pragma unroll
    for (int nf = 0; nf < 4; ++nf) bf[nf] = load_frag(sB, nf * 16, kk * 32, lane);
#pragma unroll
    for (int nf = 0; nf < 4; ++nf) {
      acc0[nf] = wmma_bf16(a0, bf[nf], acc0[nf]);
      acc1[nf] = wmma_bf16(a1, bf[nf], acc1[nf]);
    }
  }
}

// Double-buffered GEMM mainloop, buffers referenced STATICALLY (ping/pong
// phases explicit) so LDS address-space inference keeps ds_load_b128.
// Block = 128 threads (4 waves). Per wave: 32x64 output.
static __device__ __forceinline__ void gemm_mainloop(const u16* A, const u16* BT,
                                                     int m0, int n0,
                                                     u16* sA0, u16* sA1,
                                                     u16* sB0, u16* sB1,
                                                     v8f acc0[4], v8f acc1[4],
                                                     int tid, int lane, int wave) {
  const int ktiles = C_ >> 6;   // 16 (even)
  stage_tile64(sA0, A + m0 * C_, 128, C_, tid, 128);
  stage_tile64(sB0, BT + n0 * C_, 64, C_, tid, 128);
  wait_async();
  __syncthreads();
  for (int kt = 0; kt < ktiles; kt += 2) {
    // ping: stage kt+1 into buf1, compute kt from buf0
    if (kt + 1 < ktiles) {
      stage_tile64(sA1, A + m0 * C_ + (kt + 1) * 64, 128, C_, tid, 128);
      stage_tile64(sB1, BT + n0 * C_ + (kt + 1) * 64, 64, C_, tid, 128);
    }
    gemm_compute(sA0, sB0, acc0, acc1, lane, wave);
    wait_async();
    __syncthreads();
    // pong: stage kt+2 into buf0, compute kt+1 from buf1
    if (kt + 2 < ktiles) {
      stage_tile64(sA0, A + m0 * C_ + (kt + 2) * 64, 128, C_, tid, 128);
      stage_tile64(sB0, BT + n0 * C_ + (kt + 2) * 64, 64, C_, tid, 128);
    }
    gemm_compute(sA1, sB1, acc0, acc1, lane, wave);
    wait_async();
    __syncthreads();
  }
}

// ---------------------------------------------------------------- kernels ---

__global__ __launch_bounds__(256) void cvt_bf16(const float* in, u16* out, int n) {
  for (int i = blockIdx.x * blockDim.x + threadIdx.x; i < n; i += gridDim.x * blockDim.x)
    out[i] = f32_to_bf16(in[i]);
}

// out[n*K + k] = bf16(in[k*N + n])   (store W transposed, row-major [N][K])
__global__ __launch_bounds__(256) void cvt_bf16_T(const float* in, u16* out, int K, int N) {
  int total = K * N;
  for (int i = blockIdx.x * blockDim.x + threadIdx.x; i < total; i += gridDim.x * blockDim.x) {
    int k = i / N, n = i - k * N;
    out[n * K + k] = f32_to_bf16(in[i]);
  }
}

// QKV = X * W_attn^T(+b); scatter to head-major (B,H,T,D) bf16 Q/K/V buffers.
__global__ __launch_bounds__(128) void gemm_qkv(const u16* __restrict__ X,
                                                const u16* __restrict__ WT,
                                                const float* __restrict__ bias,
                                                u16* __restrict__ Q, u16* __restrict__ K,
                                                u16* __restrict__ V) {
  __shared__ __align__(16) u16 sA0[128 * LDSS], sA1[128 * LDSS];
  __shared__ __align__(16) u16 sB0[64 * LDSS], sB1[64 * LDSS];
  int tid = threadIdx.x, lane = tid & 31, wave = tid >> 5;
  int m0 = blockIdx.y * 128, n0 = blockIdx.x * 64;
  v8f z = {0.f, 0.f, 0.f, 0.f, 0.f, 0.f, 0.f, 0.f};
  v8f acc0[4] = {z, z, z, z}, acc1[4] = {z, z, z, z};
  gemm_mainloop(X, WT, m0, n0, sA0, sA1, sB0, sB1, acc0, acc1, tid, lane, wave);

  int half8 = (lane >> 4) << 3, col = lane & 15;
#pragma unroll
  for (int nf = 0; nf < 4; ++nf) {
    int n = n0 + nf * 16 + col;
    int sec = n >> 10, c = n & 1023, h = c >> 6, d = c & 63;
    u16* dst = (sec == 0) ? Q : ((sec == 1) ? K : V);
    float bv = bias[n];
#pragma unroll
    for (int ma = 0; ma < 2; ++ma) {
      const v8f& acc = ma ? acc1[nf] : acc0[nf];
#pragma unroll
      for (int r = 0; r < 8; ++r) {
        int m = m0 + wave * 32 + ma * 16 + r + half8;
        int bb = m >> 11, t = m & 2047;
        dst[((bb * H_ + h) * T_ + t) * D_ + d] = f32_to_bf16(acc[r] + bv);
      }
    }
  }
}

// Flash-style causal attention. Block = 4 waves; each wave owns 16 query rows.
__global__ __launch_bounds__(128) void attn_fa(const u16* __restrict__ Q,
                                               const u16* __restrict__ K,
                                               const u16* __restrict__ V,
                                               u16* __restrict__ O) {
  __shared__ __align__(16) u16 sQ[64 * LDSS];
  __shared__ __align__(16) u16 sK[64 * LDSS];
  __shared__ __align__(16) u16 sVt[64 * LDSS];
  __shared__ __align__(16) u16 sP[64 * LDSS];
  int tid = threadIdx.x, lane = tid & 31, wave = tid >> 5;
  int bh = blockIdx.x, qb = blockIdx.y;
  int b = bh >> 4, h = bh & 15;
  int qbase = qb * 64;
  int base = bh * T_ * D_;
  int half8 = (lane >> 4) << 3, col = lane & 15;
  const float scale = 0.125f;  // 1/sqrt(64)

  stage_tile64(sQ, Q + base + qbase * D_, 64, D_, tid, 128);
  wait_async();
  __syncthreads();
  bf16x16 qa0 = load_frag(sQ, wave * 16, 0, lane);
  bf16x16 qa1 = load_frag(sQ, wave * 16, 32, lane);

  float mrow[8], lrow[8];
  v8f z = {0.f, 0.f, 0.f, 0.f, 0.f, 0.f, 0.f, 0.f};
  v8f o[4] = {z, z, z, z};
#pragma unroll
  for (int r = 0; r < 8; ++r) { mrow[r] = -__builtin_inff(); lrow[r] = 0.f; }

  for (int kb = 0; kb <= qb; ++kb) {
    int kbase = kb * 64;
    __syncthreads();
    stage_tile64(sK, K + base + kbase * D_, 64, D_, tid, 128);
    for (int i = tid; i < 512; i += 128) {           // V transposed into LDS: sVt[d][k]
      int r = i >> 3, c0 = (i & 7) << 3;
      u16x8 v = *reinterpret_cast<const u16x8*>(V + base + kbase * D_ + r * D_ + c0);
#pragma unroll
      for (int j = 0; j < 8; ++j) sVt[(c0 + j) * LDSS + r] = v[j];
    }
    wait_async();
    __syncthreads();

    // S = Q K^T  (16x64 per wave): batch all K fragments, then WMMA burst.
    bf16x16 kf0[4], kf1[4];
#pragma unroll
    for (int nf = 0; nf < 4; ++nf) {
      kf0[nf] = load_frag(sK, nf * 16, 0, lane);
      kf1[nf] = load_frag(sK, nf * 16, 32, lane);
    }
    v8f s[4] = {z, z, z, z};
#pragma unroll
    for (int nf = 0; nf < 4; ++nf) {
      s[nf] = wmma_bf16(qa0, kf0[nf], s[nf]);
      s[nf] = wmma_bf16(qa1, kf1[nf], s[nf]);
    }

    bool diag = (kb == qb);
#pragma unroll
    for (int r = 0; r < 8; ++r) {
      int q = qbase + wave * 16 + r + half8;
      float rowm = -__builtin_inff();
#pragma unroll
      for (int nf = 0; nf < 4; ++nf) {
        int k = kbase + nf * 16 + col;
        float v = s[nf][r] * scale;
        if (diag && k > q) v = -__builtin_inff();
        s[nf][r] = v;
        rowm = fmaxf(rowm, v);
      }
#pragma unroll
      for (int off = 1; off < 16; off <<= 1) rowm = fmaxf(rowm, __shfl_xor(rowm, off, 32));
      float mn = fmaxf(mrow[r], rowm);
      float rs = 0.f;
#pragma unroll
      for (int nf = 0; nf < 4; ++nf) { float p = __expf(s[nf][r] - mn); s[nf][r] = p; rs += p; }
#pragma unroll
      for (int off = 1; off < 16; off <<= 1) rs += __shfl_xor(rs, off, 32);
      float corr = __expf(mrow[r] - mn);
      lrow[r] = lrow[r] * corr + rs;
      mrow[r] = mn;
#pragma unroll
      for (int nf = 0; nf < 4; ++nf) o[nf][r] *= corr;
#pragma unroll
      for (int nf = 0; nf < 4; ++nf)                 // C-layout -> A-layout via LDS
        sP[(wave * 16 + r + half8) * LDSS + nf * 16 + col] = f32_to_bf16(s[nf][r]);
    }
    __syncthreads();

    bf16x16 pa0 = load_frag(sP, wave * 16, 0, lane);
    bf16x16 pa1 = load_frag(sP, wave * 16, 32, lane);
    bf16x16 vf0[4], vf1[4];
#pragma unroll
    for (int df = 0; df < 4; ++df) {
      vf0[df] = load_frag(sVt, df * 16, 0, lane);
      vf1[df] = load_frag(sVt, df * 16, 32, lane);
    }
#pragma unroll
    for (int df = 0; df < 4; ++df) {                 // O += P V
      o[df] = wmma_bf16(pa0, vf0[df], o[df]);
      o[df] = wmma_bf16(pa1, vf1[df], o[df]);
    }
  }

#pragma unroll
  for (int r = 0; r < 8; ++r) {
    float inv = 1.0f / lrow[r];
    int t = qbase + wave * 16 + r + half8;
#pragma unroll
    for (int df = 0; df < 4; ++df) {
      int d = df * 16 + col;
      O[(b * T_ + t) * C_ + h * D_ + d] = f32_to_bf16(o[df][r] * inv);
    }
  }
}

// out = attn * W_proj^T + b_proj   (f32 output)
__global__ __launch_bounds__(128) void gemm_proj(const u16* __restrict__ A,
                                                 const u16* __restrict__ WT,
                                                 const float* __restrict__ bias,
                                                 float* __restrict__ out) {
  __shared__ __align__(16) u16 sA0[128 * LDSS], sA1[128 * LDSS];
  __shared__ __align__(16) u16 sB0[64 * LDSS], sB1[64 * LDSS];
  int tid = threadIdx.x, lane = tid & 31, wave = tid >> 5;
  int m0 = blockIdx.y * 128, n0 = blockIdx.x * 64;
  v8f z = {0.f, 0.f, 0.f, 0.f, 0.f, 0.f, 0.f, 0.f};
  v8f acc0[4] = {z, z, z, z}, acc1[4] = {z, z, z, z};
  gemm_mainloop(A, WT, m0, n0, sA0, sA1, sB0, sB1, acc0, acc1, tid, lane, wave);

  int half8 = (lane >> 4) << 3, col = lane & 15;
#pragma unroll
  for (int nf = 0; nf < 4; ++nf) {
    int n = n0 + nf * 16 + col;
    float bv = bias[n];
#pragma unroll
    for (int ma = 0; ma < 2; ++ma) {
      const v8f& acc = ma ? acc1[nf] : acc0[nf];
#pragma unroll
      for (int r = 0; r < 8; ++r) {
        int m = m0 + wave * 32 + ma * 16 + r + half8;
        out[m * C_ + n] = acc[r] + bv;
      }
    }
  }
}

// ------------------------------------------------------------------ launch ---

extern "C" void kernel_launch(void* const* d_in, const int* in_sizes, int n_in,
                              void* d_out, int out_size, void* d_ws, size_t ws_size,
                              hipStream_t stream) {
  (void)in_sizes; (void)n_in; (void)out_size; (void)ws_size;
  const float* x      = (const float*)d_in[0];
  const float* W_attn = (const float*)d_in[1];
  const float* b_attn = (const float*)d_in[2];
  const float* W_proj = (const float*)d_in[3];
  const float* b_proj = (const float*)d_in[4];
  float* out = (float*)d_out;

  unsigned char* w = (unsigned char*)d_ws;
  const size_t MB = 1024u * 1024u;
  u16* Xb  = (u16*)(w + 0 * MB);    // 16 MB  x in bf16
  u16* WaT = (u16*)(w + 16 * MB);   //  6 MB  W_attn^T bf16 [3072][1024]
  u16* WpT = (u16*)(w + 22 * MB);   //  2 MB  W_proj^T bf16 [1024][1024]
  u16* Qb  = (u16*)(w + 24 * MB);   // 16 MB  (B,H,T,D) bf16
  u16* Kb  = (u16*)(w + 40 * MB);   // 16 MB
  u16* Vb  = (u16*)(w + 56 * MB);   // 16 MB
  u16* Ab  = (u16*)(w + 72 * MB);   // 16 MB  attn out (B,T,C) bf16

  cvt_bf16  <<<2048, 256, 0, stream>>>(x, Xb, M_ * C_);
  cvt_bf16_T<<<2048, 256, 0, stream>>>(W_attn, WaT, C_, N3_);
  cvt_bf16_T<<<1024, 256, 0, stream>>>(W_proj, WpT, C_, C_);

  gemm_qkv <<<dim3(N3_ / 64, M_ / 128), 128, 0, stream>>>(Xb, WaT, b_attn, Qb, Kb, Vb);
  attn_fa  <<<dim3(B_ * H_, T_ / 64), 128, 0, stream>>>(Qb, Kb, Vb, Ab);
  gemm_proj<<<dim3(C_ / 64, M_ / 128), 128, 0, stream>>>(Ab, WpT, b_proj, out);
}
